// RelativeMultiHeadAttention_29781303230597
// MI455X (gfx1250) — matrix operand order repaired and checked by reading, not compiled, not run on hardware
//
#include <hip/hip_runtime.h>
#include <hip/hip_bf16.h>
#include <math.h>

// ---- problem constants (match reference) ----
#define D_MODEL  1024
#define N_HEADS  16
#define HEAD_DIM 64
#define SEQ      1024
#define BATCH    4
#define NREL     33      // 2*MAX_REL+1

typedef __attribute__((ext_vector_type(16))) __bf16 v16bf;
typedef __attribute__((ext_vector_type(8)))  float  v8f;
typedef __attribute__((ext_vector_type(8)))  __bf16 bf16x8;
typedef __attribute__((ext_vector_type(4)))  __bf16 bf16x4;
typedef __attribute__((ext_vector_type(4)))  float  f32x4;

// ---- CDNA5 async global->LDS (ASYNCcnt-tracked), one 16B move per lane ----
__device__ __forceinline__ void async_ld_b128(void* lds, const void* gaddr) {
  // LDS aperture: addr[31:0] is the LDS byte offset.
  unsigned ldsOff = (unsigned)(uintptr_t)lds;
  asm volatile("global_load_async_to_lds_b128 %0, %1, off"
               :: "v"(ldsOff), "v"((unsigned long long)(uintptr_t)gaddr)
               : "memory");
}
__device__ __forceinline__ void wait_async0() {
  asm volatile("s_wait_asynccnt 0" ::: "memory");
}

// ---- CDNA5 WMMA fragment loaders (ISA 7.12.2 layouts, wave32) ----
// A-matrix (16x32 bf16): lane L -> M = L%16; two contiguous 16B runs.
__device__ __forceinline__ v16bf load_a_frag(const __bf16* rowp, int lane) {
  const int hi8 = (lane & 16) ? 8 : 0;
  bf16x8 lo = *(const bf16x8*)(rowp + hi8);
  bf16x8 hi = *(const bf16x8*)(rowp + 16 + hi8);
  return __builtin_shufflevector(lo, hi, 0,1,2,3,4,5,6,7,8,9,10,11,12,13,14,15);
}
// B-matrix (32x16 bf16): lane L -> N = L%16; two contiguous 16B runs.
__device__ __forceinline__ v16bf load_b_frag(const __bf16* rowp, int lane) {
  const int off = (lane & 16) ? 8 : 0;
  bf16x8 lo = *(const bf16x8*)(rowp + off);
  bf16x8 hi = *(const bf16x8*)(rowp + 16 + off);
  return __builtin_shufflevector(lo, hi, 0,1,2,3,4,5,6,7,8,9,10,11,12,13,14,15);
}

__device__ __forceinline__ v8f wmma_bf16(v16bf a, v16bf b, v8f c) {
  return __builtin_amdgcn_wmma_f32_16x16x32_bf16(false, a, false, b, (short)0, c,
                                                 false, false);
}

// =====================================================================
// GEMM: out[row, n] = sum_k X[row,k] * W[n,k] + bias[n]
// Block = 128 threads (4 wave32), tile 128x64, K-stage 64.
// Each wave computes 32x64 (2 M-blocks): every B fragment feeds 2 WMMAs.
// =====================================================================
template <bool IN_BF16, bool OUT_HEAD>
__global__ __launch_bounds__(128)
void proj_gemm(const float* __restrict__ Xf, const __bf16* __restrict__ Xb,
               const float* __restrict__ W, const float* __restrict__ bias,
               __bf16* __restrict__ outH, float* __restrict__ outF)
{
  __shared__ __bf16 As[128][72];  // 144B stride: 16B aligned, non-pow2 banks
  __shared__ __bf16 Bs[64][72];

  const int tid  = threadIdx.x;
  const int lane = tid & 31;
  const int wave = tid >> 5;
  const int colBase = blockIdx.x * 64;    // N
  const int rowBase = blockIdx.y * 128;   // M

  v8f acc[2][4] = {};

  for (int k0 = 0; k0 < D_MODEL; k0 += 64) {
    // ---- stage A (128x64) ----
    if (IN_BF16) {
#pragma unroll
      for (int i = 0; i < 8; ++i) {               // 1024 x bf16x8 chunks, async
        int c8 = tid + i * 128;
        int r = c8 >> 3, c = (c8 & 7) * 8;
        async_ld_b128(&As[r][c],
                      &Xb[(size_t)(rowBase + r) * D_MODEL + k0 + c]);
      }
    } else {
#pragma unroll
      for (int i = 0; i < 16; ++i) {              // 2048 x float4 chunks
        int c4 = tid + i * 128;
        int r = c4 >> 4, c = (c4 & 15) * 4;
        f32x4 f = *(const f32x4*)&Xf[(size_t)(rowBase + r) * D_MODEL + k0 + c];
        bf16x4 o;
        o[0] = (__bf16)f[0]; o[1] = (__bf16)f[1];
        o[2] = (__bf16)f[2]; o[3] = (__bf16)f[3];
        *(bf16x4*)&As[r][c] = o;
      }
    }
    // ---- stage B = W tile (64x64), f32 -> bf16 ----
#pragma unroll
    for (int i = 0; i < 8; ++i) {
      int c4 = tid + i * 128;
      int r = c4 >> 4, c = (c4 & 15) * 4;
      f32x4 f = *(const f32x4*)&W[(size_t)(colBase + r) * D_MODEL + k0 + c];
      bf16x4 o;
      o[0] = (__bf16)f[0]; o[1] = (__bf16)f[1];
      o[2] = (__bf16)f[2]; o[3] = (__bf16)f[3];
      *(bf16x4*)&Bs[r][c] = o;
    }
    if (IN_BF16) wait_async0();
    __syncthreads();

#pragma unroll
    for (int t = 0; t < 2; ++t) {
      v16bf a0 = load_a_frag(&As[wave * 32 + (lane & 15)][t * 32], lane);
      v16bf a1 = load_a_frag(&As[wave * 32 + 16 + (lane & 15)][t * 32], lane);
#pragma unroll
      for (int j = 0; j < 4; ++j) {
        v16bf b = load_b_frag(&Bs[j * 16 + (lane & 15)][t * 32], lane);
        acc[0][j] = wmma_bf16(a0, b, acc[0][j]);
        acc[1][j] = wmma_bf16(a1, b, acc[1][j]);
      }
    }
    __syncthreads();
  }

  const int rhalf = (lane & 16) ? 8 : 0;
#pragma unroll
  for (int mb = 0; mb < 2; ++mb) {
#pragma unroll
    for (int j = 0; j < 4; ++j) {
#pragma unroll
      for (int r = 0; r < 8; ++r) {
        int rowg = rowBase + wave * 32 + mb * 16 + r + rhalf;
        int colg = colBase + j * 16 + (lane & 15);
        float v = acc[mb][j][r] + bias[colg];
        if (OUT_HEAD) {
          int b = rowg >> 10, s = rowg & (SEQ - 1);
          int h = colg >> 6,  d = colg & (HEAD_DIM - 1);
          outH[(((size_t)b * N_HEADS + h) * SEQ + s) * HEAD_DIM + d] = (__bf16)v;
        } else {
          outF[(size_t)rowg * D_MODEL + colg] = v;
        }
      }
    }
  }
}

// =====================================================================
// Relative-position dot table: p[row, j] = q[row,:] . emb_k[j,:]
// =====================================================================
__global__ void relpos_dots(const __bf16* __restrict__ qb,
                            const float* __restrict__ emb_k,
                            float* __restrict__ pbuf)
{
  int idx = blockIdx.x * blockDim.x + threadIdx.x;
  const int total = BATCH * N_HEADS * SEQ * NREL;
  if (idx >= total) return;
  int j   = idx % NREL;
  int row = idx / NREL;
  const __bf16* q = qb + (size_t)row * HEAD_DIM;
  const float*  e = emb_k + (size_t)j * HEAD_DIM;
  float s = 0.f;
#pragma unroll
  for (int d = 0; d < HEAD_DIM; ++d) s += (float)q[d] * e[d];
  pbuf[idx] = s;
}

// =====================================================================
// Flash attention with relative-position bias + bucketed emb_v term.
// One block per (b, h, 64-row query tile); 128 threads = 4 waves.
// =====================================================================
__global__ __launch_bounds__(128)
void attn_kernel(const __bf16* __restrict__ qb, const __bf16* __restrict__ kb,
                 const __bf16* __restrict__ vb, const float* __restrict__ pbuf,
                 const int* __restrict__ mask, const float* __restrict__ emb_v,
                 __bf16* __restrict__ ctxb)
{
  __shared__ __bf16 Qs[64][72];
  __shared__ __bf16 Ks[64][72];
  __shared__ __bf16 Vt[64][72];   // TRANSPOSED: Vt[dim][kcol]
  __shared__ __bf16 Ps[64][72];
  __shared__ float  SB[64][34];   // 33 relative buckets per query row (+pad)
  __shared__ float  PB[64][34];   // pbuf rows for this q-tile (kt-invariant)

  const int tid  = threadIdx.x;
  const int lane = tid & 31;
  const int wave = tid >> 5;
  const int blk  = blockIdx.x;
  const int qt = blk & 15;             // SEQ/64 = 16
  const int h  = (blk >> 4) & (N_HEADS - 1);
  const int b  = blk >> 8;
  const size_t headOff = ((size_t)b * N_HEADS + h) * SEQ * HEAD_DIM;
  const size_t pRowOff = ((size_t)b * N_HEADS + h) * SEQ;
  const int q0 = qt * 64;
  const int rhalf = (lane & 16) ? 8 : 0;

  // stage Q tile (async), pbuf rows, zero buckets
#pragma unroll
  for (int i = 0; i < 4; ++i) {
    int c8 = tid + i * 128;            // 512 x bf16x8 chunks
    int r = c8 >> 3, c = (c8 & 7) * 8;
    async_ld_b128(&Qs[r][c], &qb[headOff + (size_t)(q0 + r) * HEAD_DIM + c]);
  }
  for (int i = tid; i < 64 * NREL; i += 128) {
    int r = i / NREL, jj = i % NREL;
    PB[r][jj] = pbuf[(pRowOff + q0 + r) * NREL + jj];
    SB[r][jj] = 0.f;
  }
  wait_async0();
  __syncthreads();

  v8f oacc[4] = {};
  float m_r[8], l_r[8];
#pragma unroll
  for (int r = 0; r < 8; ++r) { m_r[r] = -1e30f; l_r[r] = 0.f; }

  for (int kt = 0; kt < SEQ / 64; ++kt) {
    const int k0 = kt * 64;
    // stage K tile (async, row-major) and V tile (transposed via VGPRs)
#pragma unroll
    for (int i = 0; i < 4; ++i) {
      int c8 = tid + i * 128;
      int r = c8 >> 3, c = (c8 & 7) * 8;
      async_ld_b128(&Ks[r][c], &kb[headOff + (size_t)(k0 + r) * HEAD_DIM + c]);
      bf16x8 vv =
          *(const bf16x8*)&vb[headOff + (size_t)(k0 + r) * HEAD_DIM + c];
#pragma unroll
      for (int e = 0; e < 8; ++e) Vt[c + e][r] = vv[e];
    }
    wait_async0();
    __syncthreads();

    // ---- scores S = Q . K^T  (16 q-rows per wave x 64 k-cols) ----
    v8f sacc[4] = {};
    const int mrow = wave * 16 + (lane & 15);
#pragma unroll
    for (int t = 0; t < 2; ++t) {          // HEAD_DIM = 2 * 32
      v16bf a = load_a_frag(&Qs[mrow][t * 32], lane);
#pragma unroll
      for (int j = 0; j < 4; ++j) {
        v16bf bfr = load_b_frag(&Ks[j * 16 + (lane & 15)][t * 32], lane);
        sacc[j] = wmma_bf16(a, bfr, sacc[j]);
      }
    }

    // ---- relative bias + mask + scale, online softmax per row ----
    float alpha[8];
#pragma unroll
    for (int r = 0; r < 8; ++r) {
      const int rowl = wave * 16 + r + rhalf;
      const int qg = q0 + rowl;
      float vals[4];
      float tmax = -1e30f;
#pragma unroll
      for (int j = 0; j < 4; ++j) {
        int kg = k0 + j * 16 + (lane & 15);
        int dd = kg - qg;
        dd = dd < -16 ? -16 : (dd > 16 ? 16 : dd);
        float v = (sacc[j][r] + PB[rowl][dd + 16]) * 0.125f;
        if (mask[((size_t)b * SEQ + qg) * SEQ + kg] == 0) v = -1e30f;
        vals[j] = v;
        tmax = fmaxf(tmax, v);
      }
#pragma unroll
      for (int off = 8; off > 0; off >>= 1)
        tmax = fmaxf(tmax, __shfl_xor(tmax, off, 32));
      float mnew = fmaxf(m_r[r], tmax);
      float al = __expf(m_r[r] - mnew);
      float rs = 0.f;
#pragma unroll
      for (int j = 0; j < 4; ++j) {
        float e = __expf(vals[j] - mnew);
        sacc[j][r] = e;
        rs += e;
      }
#pragma unroll
      for (int off = 8; off > 0; off >>= 1)
        rs += __shfl_xor(rs, off, 32);
      l_r[r] = l_r[r] * al + rs;
      m_r[r] = mnew;
      alpha[r] = al;
      oacc[0][r] *= al; oacc[1][r] *= al; oacc[2][r] *= al; oacc[3][r] *= al;
    }

    // ---- bucket rescale (one lane per half-wave owns its 8 rows) ----
    if ((lane & 15) == 0) {
#pragma unroll
      for (int r = 0; r < 8; ++r) {
        int rowl = wave * 16 + r + rhalf;
        for (int jj = 0; jj < NREL; ++jj) SB[rowl][jj] *= alpha[r];
      }
    }
    // ---- bucket accumulate + write P tile (C-layout -> LDS) ----
#pragma unroll
    for (int r = 0; r < 8; ++r) {
      const int rowl = wave * 16 + r + rhalf;
      const int qg = q0 + rowl;
#pragma unroll
      for (int j = 0; j < 4; ++j) {
        int kg = k0 + j * 16 + (lane & 15);
        int dd = kg - qg;
        dd = dd < -16 ? -16 : (dd > 16 ? 16 : dd);
        atomicAdd(&SB[rowl][dd + 16], sacc[j][r]);
        Ps[rowl][j * 16 + (lane & 15)] = (__bf16)sacc[j][r];
      }
    }

    // ---- O += P . V  (P rows wave-private; V from transposed tile) ----
#pragma unroll
    for (int t = 0; t < 2; ++t) {          // k-tile 64 = 2 * 32
      v16bf a = load_a_frag(&Ps[mrow][t * 32], lane);
#pragma unroll
      for (int j = 0; j < 4; ++j) {
        v16bf bfr = load_b_frag(&Vt[j * 16 + (lane & 15)][t * 32], lane);
        oacc[j] = wmma_bf16(a, bfr, oacc[j]);
      }
    }
    __syncthreads();
  }

  // ---- epilogue: (O + SB @ emb_v) / l -> ctx bf16 [B, S, D] ----
#pragma unroll
  for (int r = 0; r < 8; ++r) {
    const int rowl = wave * 16 + r + rhalf;
    const int qg = q0 + rowl;
    const float invl = 1.f / l_r[r];
#pragma unroll
    for (int j = 0; j < 4; ++j) {
      int dim = j * 16 + (lane & 15);
      float w2 = 0.f;
      for (int jj = 0; jj < NREL; ++jj)
        w2 += SB[rowl][jj] * emb_v[jj * HEAD_DIM + dim];
      float v = (oacc[j][r] + w2) * invl;
      ctxb[((size_t)b * SEQ + qg) * D_MODEL + h * HEAD_DIM + dim] = (__bf16)v;
    }
  }
}

// =====================================================================
extern "C" void kernel_launch(void* const* d_in, const int* in_sizes, int n_in,
                              void* d_out, int out_size, void* d_ws, size_t ws_size,
                              hipStream_t stream)
{
  (void)in_sizes; (void)n_in; (void)out_size; (void)ws_size;
  const float* query = (const float*)d_in[0];
  const float* key_  = (const float*)d_in[1];
  const float* value = (const float*)d_in[2];
  const int*   mask  = (const int*)d_in[3];
  const float* Wq = (const float*)d_in[4];
  const float* bq = (const float*)d_in[5];
  const float* Wk = (const float*)d_in[6];
  const float* bk = (const float*)d_in[7];
  const float* Wv = (const float*)d_in[8];
  const float* bv = (const float*)d_in[9];
  const float* Wo = (const float*)d_in[10];
  const float* bo = (const float*)d_in[11];
  const float* emb_k = (const float*)d_in[12];
  const float* emb_v = (const float*)d_in[13];
  float* out = (float*)d_out;

  const size_t headElems = (size_t)BATCH * N_HEADS * SEQ * HEAD_DIM; // 4,194,304
  char* ws = (char*)d_ws;
  __bf16* qb   = (__bf16*)ws;  ws += headElems * sizeof(__bf16);
  __bf16* kb   = (__bf16*)ws;  ws += headElems * sizeof(__bf16);
  __bf16* vb   = (__bf16*)ws;  ws += headElems * sizeof(__bf16);
  __bf16* ctxb = (__bf16*)ws;  ws += (size_t)BATCH * SEQ * D_MODEL * sizeof(__bf16);
  float*  pbuf = (float*)ws;   ws += (size_t)BATCH * N_HEADS * SEQ * NREL * sizeof(float);

  dim3 gGemm(D_MODEL / 64, (BATCH * SEQ) / 128);   // (16, 32)
  dim3 bGemm(128);

  // Q/K/V projections: f32 in, bf16 head-layout out
  proj_gemm<false, true><<<gGemm, bGemm, 0, stream>>>(query, nullptr, Wq, bq, qb, nullptr);
  proj_gemm<false, true><<<gGemm, bGemm, 0, stream>>>(key_,  nullptr, Wk, bk, kb, nullptr);
  proj_gemm<false, true><<<gGemm, bGemm, 0, stream>>>(value, nullptr, Wv, bv, vb, nullptr);

  // relative-position dot table
  {
    int total = BATCH * N_HEADS * SEQ * NREL;
    relpos_dots<<<(total + 255) / 256, 256, 0, stream>>>(qb, emb_k, pbuf);
  }

  // flash attention with relative bias + emb_v buckets
  attn_kernel<<<BATCH * N_HEADS * (SEQ / 64), 128, 0, stream>>>(
      qb, kb, vb, pbuf, mask, emb_v, ctxb);

  // output projection: bf16 in, f32 flat out -> d_out
  proj_gemm<true, false><<<gGemm, bGemm, 0, stream>>>(nullptr, ctxb, Wo, bo, nullptr, out);
}
